// BertGraphXLayer_13297218748519
// MI455X (gfx1250) — matrix operand order repaired and checked by reading, not compiled, not run on hardware
//
#include <hip/hip_runtime.h>
#include <hip/hip_fp16.h>

// ---------------------------------------------------------------------------
// BertGraphXLayer for MI455X (gfx1250): f16 WMMA GEMMs + WMMA attention.
// B=8 S=256 HID=768 H=12 DH=64 INTER=3072.  All dense math via
// v_wmma_f32_16x16x32_f16 (wave32).  Weights pre-transposed to [N,K] f16 so
// every LDS tile stage is a contiguous b128 copy, issued through the CDNA5
// async-to-LDS path (global_load_async_to_lds_b128 + s_wait_asynccnt).
// Graph relation bias collapsed to 2-row scalar epilogues.
// ---------------------------------------------------------------------------

#define B_     8
#define S_     256
#define H_     12
#define DH_    64
#define HID_   768
#define INTER_ 3072
#define M_     (B_ * S_)          // 2048
#define BSH_   ((size_t)M_ * HID_)
#define SCALE_ 0.125f             // 1/sqrt(64)

typedef __attribute__((ext_vector_type(16))) _Float16 v16h;
typedef __attribute__((ext_vector_type(8)))  float    v8f;

union FragH { v16h v; uint4 q[2]; };

__device__ __forceinline__ v8f zero8() {
  v8f z = {0.f, 0.f, 0.f, 0.f, 0.f, 0.f, 0.f, 0.f};
  return z;
}

__device__ __forceinline__ v8f wmma16(const v16h& a, const v16h& b, v8f c) {
  // D = A(16x32 f16) * B(32x16 f16) + C(16x16 f32)
  return __builtin_amdgcn_wmma_f32_16x16x32_f16(
      /*neg_a=*/false, a, /*neg_b=*/false, b,
      /*c_mod=*/(short)0, c, /*reuse_a=*/false, /*reuse_b=*/false);
}

// ---- CDNA5 async global->LDS staging (ASYNCcnt) via inline asm -------------
// ISA 15.18: GLOBAL_LOAD_ASYNC_TO_LDS_B128 VDST(lds addr), VADDR(64b), off.
// LDS byte address = low 32 bits of the generic LDS pointer (flat->LDS rule).
__device__ __forceinline__ void async_copy16(void* lds_dst, const void* gsrc) {
#if defined(__gfx1250__)
  asm volatile("global_load_async_to_lds_b128 %0, %1, off"
               :: "v"((unsigned)(size_t)lds_dst), "v"(gsrc)
               : "memory");
#else
  *(uint4*)lds_dst = *(const uint4*)gsrc;
#endif
}

__device__ __forceinline__ void async_wait() {
#if defined(__gfx1250__)
#if __has_builtin(__builtin_amdgcn_s_wait_asynccnt)
  __builtin_amdgcn_s_wait_asynccnt(0);
#else
  asm volatile("s_wait_asynccnt 0" ::: "memory");
#endif
#endif
}

// ---------------------------------------------------------------------------
// f32 -> f16 conversion (activations)
// ---------------------------------------------------------------------------
__global__ void cvt_f32_f16(const float* __restrict__ in,
                            _Float16* __restrict__ out, int n) {
  for (int i = blockIdx.x * 256 + threadIdx.x; i < n; i += gridDim.x * 256)
    out[i] = (_Float16)in[i];
}

// f32 W[K,N] -> f16 W^T[N,K] (weights staged once; GEMM B tiles contiguous)
__global__ void cvt_t_f32_f16(const float* __restrict__ in,
                              _Float16* __restrict__ out, int K, int N) {
  int n_elem = K * N;
  for (int i = blockIdx.x * 256 + threadIdx.x; i < n_elem; i += gridDim.x * 256) {
    int k = i / N, n = i % N;                       // coalesced read
    out[(size_t)n * K + k] = (_Float16)in[i];
  }
}

// ---------------------------------------------------------------------------
// Tiled WMMA GEMM: C[M,N] = A[M,K] * BT[N,K]^T + bias, epilogue modes:
//   0: f32 out   1: f16 out   2: f16 out with exact GELU
//   3: f16 out transposed as [B,H,DH,S]  (for V tensors)
// Block = 128 threads (4 waves, 2x2 wave grid), tile 64x64, K-step 32.
// ---------------------------------------------------------------------------
__global__ __launch_bounds__(128)
void gemm_f16_wmma(const _Float16* __restrict__ A, const _Float16* __restrict__ BT,
                   const float* __restrict__ bias, void* __restrict__ out,
                   int K, int N, int mode) {
  __shared__ __align__(16) _Float16 sA[64 * 32];   // [m][k]
  __shared__ __align__(16) _Float16 sB[64 * 32];   // [n][k]

  const int tid  = threadIdx.x;
  const int lane = tid & 31, wave = tid >> 5;
  const int g    = lane >> 4, ln = lane & 15;
  const int wm   = wave >> 1, wn = wave & 1;
  const int m0   = blockIdx.x * 64, n0 = blockIdx.y * 64;

  v8f acc[2][2];
  acc[0][0] = zero8(); acc[0][1] = zero8();
  acc[1][0] = zero8(); acc[1][1] = zero8();

  for (int k0 = 0; k0 < K; k0 += 32) {
    __syncthreads();
#pragma unroll
    for (int j = 0; j < 2; ++j) {            // A tile: 256 x 16B chunks
      int ch = tid + j * 128;
      int row = ch >> 2, cc = (ch & 3) * 8;
      async_copy16(&sA[row * 32 + cc], &A[(size_t)(m0 + row) * K + k0 + cc]);
    }
#pragma unroll
    for (int j = 0; j < 2; ++j) {            // B tile (pre-transposed weights)
      int ch = tid + j * 128;
      int nr = ch >> 2, cc = (ch & 3) * 8;
      async_copy16(&sB[nr * 32 + cc], &BT[(size_t)(n0 + nr) * K + k0 + cc]);
    }
    if (k0 + 32 < K) {                       // prefetch next K-step tiles
      int ch = tid;
      int row = ch >> 2, cc = (ch & 3) * 8;
      __builtin_prefetch(&A[(size_t)(m0 + row) * K + k0 + 32 + cc], 0, 3);
      __builtin_prefetch(&BT[(size_t)(n0 + row) * K + k0 + 32 + cc], 0, 3);
    }
    async_wait();
    __syncthreads();

    FragH fa[2], fb[2];
#pragma unroll
    for (int tm = 0; tm < 2; ++tm) {         // A frag: K = 16*(i/8)+8g+(i%8)
      const _Float16* p = &sA[(wm * 32 + tm * 16 + ln) * 32];
      fa[tm].q[0] = *(const uint4*)(p + 8 * g);
      fa[tm].q[1] = *(const uint4*)(p + 16 + 8 * g);
    }
#pragma unroll
    for (int tn = 0; tn < 2; ++tn) {         // B frag: K = 16*(i/8)+8g+(i%8)
      const _Float16* p = &sB[(wn * 32 + tn * 16 + ln) * 32];
      fb[tn].q[0] = *(const uint4*)(p + 8 * g);
      fb[tn].q[1] = *(const uint4*)(p + 16 + 8 * g);
    }
#pragma unroll
    for (int tm = 0; tm < 2; ++tm)
#pragma unroll
      for (int tn = 0; tn < 2; ++tn)
        acc[tm][tn] = wmma16(fa[tm].v, fb[tn].v, acc[tm][tn]);
  }

#pragma unroll
  for (int tm = 0; tm < 2; ++tm)
#pragma unroll
    for (int tn = 0; tn < 2; ++tn) {
      int col = n0 + wn * 32 + tn * 16 + ln;
      float bv = bias[col];
#pragma unroll
      for (int r = 0; r < 8; ++r) {          // C layout: M = r + 8g, N = ln
        int row = m0 + wm * 32 + tm * 16 + r + 8 * g;
        float v = acc[tm][tn][r] + bv;
        if (mode == 2) v = 0.5f * v * (1.f + erff(v * 0.70710678118654752f));
        if (mode == 0) {
          ((float*)out)[(size_t)row * N + col] = v;
        } else if (mode == 3) {              // V -> [B,H,DH,S] f16
          int bb = row >> 8, s = row & (S_ - 1);
          int hh = col >> 6, d = col & 63;
          ((_Float16*)out)[(((size_t)bb * H_ + hh) * DH_ + d) * S_ + s] = (_Float16)v;
        } else {
          ((_Float16*)out)[(size_t)row * N + col] = (_Float16)v;
        }
      }
    }
}

// ---------------------------------------------------------------------------
// Attention: block = 4 waves, handles (b, h, 64 q-rows); each wave owns 16
// q-rows. Scores & ctx via WMMA; softmax + graph bias via lane shuffles.
// K [B,S,HID] and Vt [B,H,DH,S] staged through one shared 128-key chunk
// buffer via async-to-LDS (contiguous b128 chunks).
// ---------------------------------------------------------------------------
__global__ __launch_bounds__(128)
void attn_wmma(const _Float16* __restrict__ Qm, const _Float16* __restrict__ Km,
               const _Float16* __restrict__ Vt, const float* __restrict__ mask,
               const int* __restrict__ graph, const float* __restrict__ dpk_ln,
               const float* __restrict__ dpv_ln,
               _Float16* __restrict__ ctx16, float* __restrict__ ctx32) {
  __shared__ __align__(16) _Float16 sKV[128 * 64];      // K: [s][d]; V: [d][s]
  __shared__ __align__(16) _Float16 sP[4][16 * 128];    // per-wave probs chunk
  __shared__ float sS0[4][16], sS1[4][16];

  const int tid  = threadIdx.x;
  const int lane = tid & 31, wave = tid >> 5;
  const int g    = lane >> 4, ln = lane & 15;
  const int b    = blockIdx.x / H_;
  const int h    = blockIdx.x % H_;
  const int qw   = blockIdx.y * 64 + wave * 16;

  // relation-score biases s0/s1 per q-row (q . LN(dpk[j]))
  if (graph != nullptr && lane < 16) {
    const _Float16* qp = &Qm[((size_t)(b * S_ + qw + lane)) * HID_ + h * DH_];
    float a0 = 0.f, a1 = 0.f;
    for (int d = 0; d < DH_; ++d) {
      float qv = (float)qp[d];
      a0 += qv * dpk_ln[d];
      a1 += qv * dpk_ln[DH_ + d];
    }
    sS0[wave][lane] = a0; sS1[wave][lane] = a1;
  }

  // Q fragments (A-matrix, K-dim 64 -> 2 frags), straight from global
  FragH qa[2];
  {
    const _Float16* qp = &Qm[((size_t)(b * S_ + qw + ln)) * HID_ + h * DH_];
    qa[0].q[0] = *(const uint4*)(qp + 8 * g);
    qa[0].q[1] = *(const uint4*)(qp + 16 + 8 * g);
    qa[1].q[0] = *(const uint4*)(qp + 32 + 8 * g);
    qa[1].q[1] = *(const uint4*)(qp + 48 + 8 * g);
  }

  v8f sc[16];                                // 16 score tiles = 256 keys

  // ---- scores = Q K^T ----
  for (int c = 0; c < 2; ++c) {
    __syncthreads();
    for (int ch = tid; ch < 1024; ch += 128) {       // K chunk -> [s][d]
      int s = ch >> 3, dc = (ch & 7) * 8;
      async_copy16(&sKV[s * 64 + dc],
                   &Km[((size_t)(b * S_ + c * 128 + s)) * HID_ + h * DH_ + dc]);
    }
    async_wait();
    __syncthreads();
#pragma unroll
    for (int tl = 0; tl < 8; ++tl) {
      const _Float16* kp = &sKV[(tl * 16 + ln) * 64];
      FragH kb0, kb1;                                // B frag: K(=d) = 16*(i/8)+8g+(i%8)
      kb0.q[0] = *(const uint4*)(kp + 8 * g);      kb0.q[1] = *(const uint4*)(kp + 16 + 8 * g);
      kb1.q[0] = *(const uint4*)(kp + 32 + 8 * g); kb1.q[1] = *(const uint4*)(kp + 48 + 8 * g);
      v8f z = zero8();
      z = wmma16(qa[0].v, kb0.v, z);
      z = wmma16(qa[1].v, kb1.v, z);
      sc[c * 8 + tl] = z;
    }
  }

  // ---- epilogue: scale + mask + graph bias; remember arc==0 bits ----
  unsigned arcb[4] = {0u, 0u, 0u, 0u};
#pragma unroll
  for (int t = 0; t < 16; ++t) {
    int key = t * 16 + ln;
    float mk = mask[b * S_ + key];
#pragma unroll
    for (int r = 0; r < 8; ++r) {
      float v = sc[t][r] * SCALE_ + mk;
      if (graph != nullptr) {
        int m = r + 8 * g;
        int arc = graph[((size_t)(b * S_ + qw + m)) * S_ + key];
        v += (arc != 0) ? sS1[wave][m] : sS0[wave][m];
        if (arc == 0) { int bi = t * 8 + r; arcb[bi >> 5] |= (1u << (bi & 31)); }
      }
      sc[t][r] = v;
    }
  }

  // ---- softmax per row (row m = r + 8g lives on 16 lanes of group g) ----
  float w0[8];
#pragma unroll
  for (int r = 0; r < 8; ++r) {
    float mx = -3.0e38f;
#pragma unroll
    for (int t = 0; t < 16; ++t) mx = fmaxf(mx, sc[t][r]);
    for (int off = 1; off < 16; off <<= 1) mx = fmaxf(mx, __shfl_xor(mx, off, 32));
    float sm = 0.f;
#pragma unroll
    for (int t = 0; t < 16; ++t) { float e = __expf(sc[t][r] - mx); sc[t][r] = e; sm += e; }
    for (int off = 1; off < 16; off <<= 1) sm += __shfl_xor(sm, off, 32);
    float inv = 1.f / sm;
    float a0 = 0.f;
#pragma unroll
    for (int t = 0; t < 16; ++t) {
      float p = sc[t][r] * inv;
      sc[t][r] = p;
      int bi = t * 8 + r;
      if ((arcb[bi >> 5] >> (bi & 31)) & 1u) a0 += p;
    }
    for (int off = 1; off < 16; off <<= 1) a0 += __shfl_xor(a0, off, 32);
    w0[r] = a0;                                    // sum of probs where arc==0
  }

  // ---- ctx = P V ----
  v8f cacc[4];
#pragma unroll
  for (int dt = 0; dt < 4; ++dt) cacc[dt] = zero8();

  for (int c = 0; c < 2; ++c) {
    __syncthreads();
    for (int ch = tid; ch < 1024; ch += 128) {       // V chunk -> [d][s_loc]
      int d = ch >> 4, sc8 = (ch & 15) * 8;
      async_copy16(&sKV[d * 128 + sc8],
                   &Vt[(((size_t)b * H_ + h) * DH_ + d) * S_ + c * 128 + sc8]);
    }
    _Float16* pw = &sP[wave][0];                     // probs -> A layout via LDS
#pragma unroll
    for (int tl = 0; tl < 8; ++tl)
#pragma unroll
      for (int r = 0; r < 8; ++r)
        pw[(r + 8 * g) * 128 + tl * 16 + ln] = (_Float16)sc[c * 8 + tl][r];
    async_wait();
    __syncthreads();
#pragma unroll
    for (int u = 0; u < 4; ++u) {
      FragH pa;
      const _Float16* pp = &pw[ln * 128 + u * 32];
      pa.q[0] = *(const uint4*)(pp + 8 * g);
      pa.q[1] = *(const uint4*)(pp + 16 + 8 * g);
#pragma unroll
      for (int dt = 0; dt < 4; ++dt) {
        FragH vb;
        const _Float16* vp = &sKV[(dt * 16 + ln) * 128 + u * 32];
        vb.q[0] = *(const uint4*)(vp + 8 * g);
        vb.q[1] = *(const uint4*)(vp + 16 + 8 * g);
        cacc[dt] = wmma16(pa.v, vb.v, cacc[dt]);
      }
    }
  }

  // ---- ctx epilogue: dpv mixing + store ----
#pragma unroll
  for (int dt = 0; dt < 4; ++dt) {
    int d = dt * 16 + ln;
    float dv0 = 0.f, dv1 = 0.f;
    if (graph != nullptr) { dv0 = dpv_ln[d]; dv1 = dpv_ln[DH_ + d]; }
#pragma unroll
    for (int r = 0; r < 8; ++r) {
      int m = r + 8 * g;
      float v = cacc[dt][r];
      if (graph != nullptr) v += dv0 * w0[r] + dv1 * (1.f - w0[r]);
      size_t o = ((size_t)(b * S_ + qw + m)) * HID_ + h * DH_ + d;
      ctx16[o] = (_Float16)v;
      if (ctx32) ctx32[o] = v;
    }
  }
}

// ---------------------------------------------------------------------------
// Fused residual + LayerNorm over HID=768; block-per-row, 256 threads.
// ---------------------------------------------------------------------------
__global__ __launch_bounds__(256)
void ln_res(const float* __restrict__ hsrc, const float* __restrict__ res,
            const float* __restrict__ gamma, const float* __restrict__ beta,
            float eps, float* __restrict__ out32, _Float16* __restrict__ out16) {
  const int row = blockIdx.x, tid = threadIdx.x;
  __shared__ float red[8];
  float x[3];
  float s = 0.f;
#pragma unroll
  for (int j = 0; j < 3; ++j) {
    int col = j * 256 + tid;
    x[j] = hsrc[(size_t)row * HID_ + col] + res[(size_t)row * HID_ + col];
    s += x[j];
  }
  for (int off = 1; off < 32; off <<= 1) s += __shfl_xor(s, off, 32);
  if ((tid & 31) == 0) red[tid >> 5] = s;
  __syncthreads();
  float tot = 0.f;
  for (int w = 0; w < 8; ++w) tot += red[w];
  float mu = tot * (1.f / (float)HID_);
  __syncthreads();
  float s2 = 0.f;
#pragma unroll
  for (int j = 0; j < 3; ++j) { float d = x[j] - mu; s2 += d * d; }
  for (int off = 1; off < 32; off <<= 1) s2 += __shfl_xor(s2, off, 32);
  if ((tid & 31) == 0) red[tid >> 5] = s2;
  __syncthreads();
  float tot2 = 0.f;
  for (int w = 0; w < 8; ++w) tot2 += red[w];
  float rstd = rsqrtf(tot2 * (1.f / (float)HID_) + eps);
#pragma unroll
  for (int j = 0; j < 3; ++j) {
    int col = j * 256 + tid;
    float y = (x[j] - mu) * rstd * gamma[col] + beta[col];
    if (out32) out32[(size_t)row * HID_ + col] = y;
    if (out16) out16[(size_t)row * HID_ + col] = (_Float16)y;
  }
}

// LayerNorm of the 2x64 relation embeddings (eps = 1e-5), grid = 2 rows.
__global__ void ln_small(const float* __restrict__ src, const float* __restrict__ gamma,
                         const float* __restrict__ beta, float* __restrict__ dst) {
  __shared__ float buf[64];
  __shared__ float stats[2];
  int r = blockIdx.x, t = threadIdx.x;
  buf[t] = src[r * 64 + t];
  __syncthreads();
  if (t == 0) {
    float s = 0.f;
    for (int i = 0; i < 64; ++i) s += buf[i];
    float mu = s * (1.f / 64.f);
    float v = 0.f;
    for (int i = 0; i < 64; ++i) { float d = buf[i] - mu; v += d * d; }
    stats[0] = mu; stats[1] = rsqrtf(v * (1.f / 64.f) + 1e-5f);
  }
  __syncthreads();
  dst[r * 64 + t] = (buf[t] - stats[0]) * stats[1] * gamma[t] + beta[t];
}

// ---------------------------------------------------------------------------
// Host driver
// ---------------------------------------------------------------------------
extern "C" void kernel_launch(void* const* d_in, const int* in_sizes, int n_in,
                              void* d_out, int out_size, void* d_ws, size_t ws_size,
                              hipStream_t stream) {
  (void)in_sizes; (void)n_in; (void)out_size; (void)ws_size;

  const float* lang_feats = (const float*)d_in[0];
  const float* visn_feats = (const float*)d_in[1];
  const float* lang_mask  = (const float*)d_in[2];
  const float* visn_mask  = (const float*)d_in[3];
  const int*   arc_lang   = (const int*)d_in[4];
  const int*   arc_visn   = (const int*)d_in[5];
  auto P = [&](int i) { return (const float*)d_in[i]; };
  // params (insertion order): 6..23 qkv c/l/v, 24..27 c_out/ln, 28.. l block, 40.. v block

  char* ws = (char*)d_ws;
  size_t off = 0;
  auto a16 = [&](size_t n) { _Float16* p = (_Float16*)(ws + off);
                             off = (off + n * 2 + 255) & ~(size_t)255; return p; };
  auto a32 = [&](size_t n) { float* p = (float*)(ws + off);
                             off = (off + n * 4 + 255) & ~(size_t)255; return p; };

  _Float16 *x16l = a16(BSH_), *x16v = a16(BSH_);
  _Float16 *wQc = a16(HID_*HID_), *wKc = a16(HID_*HID_), *wVc = a16(HID_*HID_);
  _Float16 *wQl = a16(HID_*HID_), *wKl = a16(HID_*HID_), *wVl = a16(HID_*HID_);
  _Float16 *wQv = a16(HID_*HID_), *wKv = a16(HID_*HID_), *wVv = a16(HID_*HID_);
  _Float16 *wCo = a16(HID_*HID_);
  _Float16 *wFiL = a16((size_t)HID_*INTER_), *wFoL = a16((size_t)INTER_*HID_);
  _Float16 *wFiV = a16((size_t)HID_*INTER_), *wFoV = a16((size_t)INTER_*HID_);
  _Float16 *q16[6]; for (int i = 0; i < 6; ++i) q16[i] = a16(BSH_);
  _Float16 *ctxA = a16(BSH_), *ctxB = a16(BSH_);
  _Float16 *hi16 = a16((size_t)M_ * INTER_);
  float *h32a = a32(BSH_), *h32b = a32(BSH_);
  float *s32a = a32(BSH_), *s32b = a32(BSH_);
  float *dpln = a32(512);   // [l_dpk(2x64) | l_dpv | v_dpk | v_dpv]

  // ---- activations -> f16 ----
  cvt_f32_f16<<<dim3(512), dim3(256), 0, stream>>>(lang_feats, x16l, (int)BSH_);
  cvt_f32_f16<<<dim3(512), dim3(256), 0, stream>>>(visn_feats, x16v, (int)BSH_);

  // ---- weights -> f16 transposed [N,K] ----
  struct CvtT { const float* s; _Float16* d; int K, N; } cvt[] = {
    {P(6),  wQc, HID_, HID_}, {P(8),  wKc, HID_, HID_}, {P(10), wVc, HID_, HID_},
    {P(12), wQl, HID_, HID_}, {P(14), wKl, HID_, HID_}, {P(16), wVl, HID_, HID_},
    {P(18), wQv, HID_, HID_}, {P(20), wKv, HID_, HID_}, {P(22), wVv, HID_, HID_},
    {P(24), wCo, HID_, HID_},
    {P(34), wFiL, HID_, INTER_}, {P(36), wFoL, INTER_, HID_},
    {P(46), wFiV, HID_, INTER_}, {P(48), wFoV, INTER_, HID_},
  };
  for (auto& c : cvt)
    cvt_t_f32_f16<<<dim3(512), dim3(256), 0, stream>>>(c.s, c.d, c.K, c.N);

  // ---- layernorm the 2x64 relation embeddings ----
  ln_small<<<dim3(2), dim3(64), 0, stream>>>(P(28), P(30), P(31), dpln +   0); // l_dpk
  ln_small<<<dim3(2), dim3(64), 0, stream>>>(P(29), P(32), P(33), dpln + 128); // l_dpv
  ln_small<<<dim3(2), dim3(64), 0, stream>>>(P(40), P(42), P(43), dpln + 256); // v_dpk
  ln_small<<<dim3(2), dim3(64), 0, stream>>>(P(41), P(44), P(45), dpln + 384); // v_dpv

  dim3 gg(M_ / 64, HID_ / 64), gb(128);
  dim3 gi(M_ / 64, INTER_ / 64);
  dim3 ga(B_ * H_, S_ / 64), ab(128);

  // ---- cross-attention QKV (shared weights, both streams); V transposed ----
  gemm_f16_wmma<<<gg, gb, 0, stream>>>(x16l, wQc, P(7),  q16[0], HID_, HID_, 1);
  gemm_f16_wmma<<<gg, gb, 0, stream>>>(x16l, wKc, P(9),  q16[1], HID_, HID_, 1);
  gemm_f16_wmma<<<gg, gb, 0, stream>>>(x16l, wVc, P(11), q16[2], HID_, HID_, 3);
  gemm_f16_wmma<<<gg, gb, 0, stream>>>(x16v, wQc, P(7),  q16[3], HID_, HID_, 1);
  gemm_f16_wmma<<<gg, gb, 0, stream>>>(x16v, wKc, P(9),  q16[4], HID_, HID_, 1);
  gemm_f16_wmma<<<gg, gb, 0, stream>>>(x16v, wVc, P(11), q16[5], HID_, HID_, 3);

  // lang attends visn; visn attends lang
  attn_wmma<<<ga, ab, 0, stream>>>(q16[0], q16[4], q16[5], visn_mask,
                                   nullptr, nullptr, nullptr, ctxA, nullptr);
  attn_wmma<<<ga, ab, 0, stream>>>(q16[3], q16[1], q16[2], lang_mask,
                                   nullptr, nullptr, nullptr, ctxB, nullptr);

  // out-projection + residual LN (BertAttOutput, eps 1e-12) -> lang_x / visn_x
  gemm_f16_wmma<<<gg, gb, 0, stream>>>(ctxA, wCo, P(25), h32a, HID_, HID_, 0);
  gemm_f16_wmma<<<gg, gb, 0, stream>>>(ctxB, wCo, P(25), h32b, HID_, HID_, 0);
  ln_res<<<dim3(M_), dim3(256), 0, stream>>>(h32a, lang_feats, P(26), P(27),
                                             1e-12f, nullptr, x16l);
  ln_res<<<dim3(M_), dim3(256), 0, stream>>>(h32b, visn_feats, P(26), P(27),
                                             1e-12f, nullptr, x16v);

  // ---- graph-biased self-attention ----
  gemm_f16_wmma<<<gg, gb, 0, stream>>>(x16l, wQl, P(13), q16[0], HID_, HID_, 1);
  gemm_f16_wmma<<<gg, gb, 0, stream>>>(x16l, wKl, P(15), q16[1], HID_, HID_, 1);
  gemm_f16_wmma<<<gg, gb, 0, stream>>>(x16l, wVl, P(17), q16[2], HID_, HID_, 3);
  gemm_f16_wmma<<<gg, gb, 0, stream>>>(x16v, wQv, P(19), q16[3], HID_, HID_, 1);
  gemm_f16_wmma<<<gg, gb, 0, stream>>>(x16v, wKv, P(21), q16[4], HID_, HID_, 1);
  gemm_f16_wmma<<<gg, gb, 0, stream>>>(x16v, wVv, P(23), q16[5], HID_, HID_, 3);

  attn_wmma<<<ga, ab, 0, stream>>>(q16[0], q16[1], q16[2], lang_mask,
                                   arc_lang, dpln + 0, dpln + 128, ctxA, s32a);
  attn_wmma<<<ga, ab, 0, stream>>>(q16[3], q16[4], q16[5], visn_mask,
                                   arc_visn, dpln + 256, dpln + 384, ctxB, s32b);

  // ---- FFN (GELU intermediate) + residual LN -> d_out ----
  float* outF = (float*)d_out;
  gemm_f16_wmma<<<gi, gb, 0, stream>>>(ctxA, wFiL, P(35), hi16, HID_, INTER_, 2);
  gemm_f16_wmma<<<gg, gb, 0, stream>>>(hi16, wFoL, P(37), h32a, INTER_, HID_, 0);
  ln_res<<<dim3(M_), dim3(256), 0, stream>>>(h32a, s32a, P(38), P(39),
                                             1e-12f, outF, nullptr);

  gemm_f16_wmma<<<gi, gb, 0, stream>>>(ctxB, wFiV, P(47), hi16, HID_, INTER_, 2);
  gemm_f16_wmma<<<gg, gb, 0, stream>>>(hi16, wFoV, P(49), h32b, INTER_, HID_, 0);
  ln_res<<<dim3(M_), dim3(256), 0, stream>>>(h32b, s32b, P(50), P(51),
                                             1e-12f, outF + BSH_, nullptr);
}
